// MultiHeadedAttention_39608188403850
// MI455X (gfx1250) — matrix operand order repaired and checked by reading, not compiled, not run on hardware
//
#include <hip/hip_runtime.h>
#include <hip/hip_bf16.h>

// ---------------------------------------------------------------------------
// MultiHeadedAttention on MI455X (gfx1250, wave32, WMMA)
//   B=2, L=2048, HID=1024, NH=16, DH=64
// Pipeline:
//   gemm<f32 in, f16 out> x3  : Qp/Kp/Vp projections (f16 in scratch)
//   attn_flash_wmma           : flash attention, f16 tiles, f32 accum,
//                               writes Ctx f16 at (b, l, h*64+d)
//   gemm<f16 in, f32 out>     : out = Ctx @ Wo + bo
// Faithful reshape: head h of batch b is the contiguous (L,64) slice at
//   P + b*L*1024 + h*L*64   (raw .view without transpose)
// ---------------------------------------------------------------------------

typedef _Float16 v16h __attribute__((ext_vector_type(16)));
typedef _Float16 v8h  __attribute__((ext_vector_type(8)));
typedef float    v8f  __attribute__((ext_vector_type(8)));

#define ATT_B   2
#define ATT_L   2048
#define ATT_HID 1024
#define ATT_NH  16
#define ATT_DH  64
#define ATT_M   (ATT_B * ATT_L)   // 4096 rows for the projection GEMMs
// All four GEMMs are M=4096, N=1024, K=1024 (compile-time -> shift addressing)
#define GM 4096
#define GN 1024
#define GK 1024

static __device__ __forceinline__ v16h hcombine(v8h lo, v8h hi) {
  union { v16h v; v8h h[2]; } u;
  u.h[0] = lo; u.h[1] = hi;
  return u.v;
}

static __device__ __forceinline__ v8f wmma_f16(v16h a, v16h b, v8f c) {
  // v_wmma_f32_16x16x32_f16  (8-arg form, probe-confirmed)
  return __builtin_amdgcn_wmma_f32_16x16x32_f16(
      /*neg_a=*/false, a, /*neg_b=*/false, b,
      /*c_mod=*/(short)0, c, /*reuse_a=*/false, /*reuse_b=*/false);
}

// ---------------------------------------------------------------------------
// GEMM:  C[4096,1024] = A[4096,1024] @ W[1024,1024] + bias[1024]
//   A_IS_F16 = false : A is f32 (raw inputs), staged with cvt
//   A_IS_F16 = true  : A is f16 (Ctx scratch), staged as pure b128 copy
//   OutT = _Float16 (projection scratch) or float (final output)
// 128x128 tile per 256-thread block (8 waves); WMMA 16x16x32 f16, f32 accum.
// LDS pitch = 40 halves = 80 B (multiple of 16 B -> aligned ds_load_b128).
// ---------------------------------------------------------------------------
template <typename OutT, bool A_IS_F16>
__global__ __launch_bounds__(256)
void gemm_bias_wmma(const void* __restrict__ Av, const float* __restrict__ W,
                    const float* __restrict__ bias, OutT* __restrict__ C) {
  __shared__ _Float16 Alds[128 * 40];   // [m][k]  row-major
  __shared__ _Float16 Btlds[128 * 40];  // [n][k]  (W tile transposed)

  const int tid  = threadIdx.x;
  const int lane = tid & 31;
  const int wave = tid >> 5;     // 0..7
  const int wm   = wave >> 2;    // 0..1 : 64-row slab
  const int wn   = wave & 3;     // 0..3 : 32-col slab
  const int l16  = lane & 15;
  const int hi   = lane >> 4;    // half-wave select
  const int m0   = blockIdx.x * 128;
  const int n0   = blockIdx.y * 128;

  v8f acc[4][2] = {};

  for (int k0 = 0; k0 < GK; k0 += 32) {
    // ---- stage A tile (128x32 -> f16 LDS) ----
    if constexpr (A_IS_F16) {
      const _Float16* A = (const _Float16*)Av;
      const int r = tid >> 2;            // 0..63
      const int c = (tid & 3) * 8;       // 0,8,16,24
#pragma unroll
      for (int i = 0; i < 2; ++i) {
        const int row = r + i * 64;
        *(v8h*)&Alds[row * 40 + c] =
            *(const v8h*)(A + (size_t)(m0 + row) * GK + k0 + c);
      }
    } else {
      const float* A = (const float*)Av;
      const int r = tid >> 3;            // 0..31
      const int c = (tid & 7) * 4;       // 0..28
#pragma unroll
      for (int i = 0; i < 4; ++i) {
        const int row = r + i * 32;
        const float4 f =
            *(const float4*)(A + (size_t)(m0 + row) * GK + k0 + c);
        _Float16* d = &Alds[row * 40 + c];
        d[0] = (_Float16)f.x; d[1] = (_Float16)f.y;
        d[2] = (_Float16)f.z; d[3] = (_Float16)f.w;
      }
    }
    // ---- stage W tile (32x128 f32 -> f16, transposed) ----
    {
      const int r = tid >> 5;            // 0..7
      const int c = (tid & 31) * 4;      // 0..124
#pragma unroll
      for (int i = 0; i < 4; ++i) {
        const int kk = r + i * 8;        // 0..31
        const float4 f =
            *(const float4*)(W + (size_t)(k0 + kk) * GN + n0 + c);
        Btlds[(c + 0) * 40 + kk] = (_Float16)f.x;
        Btlds[(c + 1) * 40 + kk] = (_Float16)f.y;
        Btlds[(c + 2) * 40 + kk] = (_Float16)f.z;
        Btlds[(c + 3) * 40 + kk] = (_Float16)f.w;
      }
    }
    __syncthreads();

    // ---- B fragments for this wave's two 16-col subtiles ----
    v16h bfrag[2];
#pragma unroll
    for (int ni = 0; ni < 2; ++ni) {
      const int n  = wn * 32 + ni * 16 + l16;
      const int kb = hi * 8;   // halves 0-7 = K kb..kb+7, 8-15 = +16
      bfrag[ni] = hcombine(*(const v8h*)&Btlds[n * 40 + kb],
                           *(const v8h*)&Btlds[n * 40 + kb + 16]);
    }
    // ---- A fragments + 8 WMMAs ----
#pragma unroll
    for (int mi = 0; mi < 4; ++mi) {
      const int m  = wm * 64 + mi * 16 + l16;
      const int kb = hi * 8;   // A layout: halves 0-7 = K kb..kb+7, 8-15 = +16
      const v16h afrag = hcombine(*(const v8h*)&Alds[m * 40 + kb],
                                  *(const v8h*)&Alds[m * 40 + kb + 16]);
#pragma unroll
      for (int ni = 0; ni < 2; ++ni)
        acc[mi][ni] = wmma_f16(afrag, bfrag[ni], acc[mi][ni]);
    }
    __syncthreads();
  }

  // ---- epilogue: C = acc + bias (compile-time strides -> shift math) ----
#pragma unroll
  for (int ni = 0; ni < 2; ++ni) {
    const int n = n0 + wn * 32 + ni * 16 + l16;
    const float bn = bias[n];
#pragma unroll
    for (int mi = 0; mi < 4; ++mi) {
      const int mb = m0 + wm * 64 + mi * 16 + hi * 8;
      OutT* cp = C + (size_t)mb * GN + n;
#pragma unroll
      for (int r = 0; r < 8; ++r)
        cp[(size_t)r * GN] = (OutT)(acc[mi][ni][r] + bn);
    }
  }
}

// ---------------------------------------------------------------------------
// Flash attention per (q-tile, head, batch).  Block = 128 threads = 4 waves,
// each wave owns 16 q rows.  Q/K/V already f16 -> staging is pure b128 copy.
// Online softmax stats replicated across 16-lane groups via shfl_xor (wave32).
// Ctx (f16) written at (b, l, h*64+d)  (the transpose(1,2).reshape of ref).
// ---------------------------------------------------------------------------
__global__ __launch_bounds__(128)
void attn_flash_wmma(const _Float16* __restrict__ Qp,
                     const _Float16* __restrict__ Kp,
                     const _Float16* __restrict__ Vp,
                     const int* __restrict__ Mask,
                     _Float16* __restrict__ Ctx) {
  __shared__ _Float16 Qlds[64 * 72];        //  9216 B  [qrow][d]
  __shared__ _Float16 Klds[128 * 72];       // 18432 B  [key][d]
  __shared__ _Float16 Vtlds[64 * 136];      // 17408 B  [d][key]
  __shared__ _Float16 Plds[4 * 16 * 136];   // 17408 B  per-wave [row][key]

  const int b  = blockIdx.z;
  const int h  = blockIdx.y;
  const int q0 = blockIdx.x * 64;
  const int tid  = threadIdx.x;
  const int lane = tid & 31;
  const int wave = tid >> 5;   // 0..3
  const int l16  = lane & 15;
  const int hi   = lane >> 4;

  const size_t headoff = ((size_t)b * ATT_NH + h) * (size_t)ATT_L * ATT_DH;
  const _Float16* Qh = Qp + headoff;       // (L, 64) row-major, contiguous
  const _Float16* Kh = Kp + headoff;
  const _Float16* Vh = Vp + headoff;
  const int*      Mb = Mask + (size_t)b * ATT_L * ATT_L;

  // ---- stage Q tile (64x64 halves) once: pure b128 copy ----
  for (int idx = tid; idx < 64 * 8; idx += 128) {
    const int row = idx >> 3;
    const int c   = (idx & 7) * 8;
    *(v8h*)&Qlds[row * 72 + c] =
        *(const v8h*)(Qh + (size_t)(q0 + row) * ATT_DH + c);
  }
  __syncthreads();

  // loop-invariant Q fragments (d = 0..31, 32..63)
  v16h qf[2];
  {
    const int m = wave * 16 + l16;
#pragma unroll
    for (int ds = 0; ds < 2; ++ds) {
      const int kb = ds * 32 + hi * 8;
      qf[ds] = hcombine(*(const v8h*)&Qlds[m * 72 + kb],
                        *(const v8h*)&Qlds[m * 72 + kb + 16]);
    }
  }

  v8f ctx[4] = {};
  float rmax[8], rsum[8];
#pragma unroll
  for (int r = 0; r < 8; ++r) { rmax[r] = -3.0e38f; rsum[r] = 0.0f; }

  for (int kb0 = 0; kb0 < ATT_L; kb0 += 128) {
    __syncthreads();  // previous iteration done with Klds / Vtlds

    if (kb0 + 128 < ATT_L) {   // gfx1250 global_prefetch_b8 path
      __builtin_prefetch(Kh + (size_t)(kb0 + 128) * ATT_DH, 0, 1);
      __builtin_prefetch(Vh + (size_t)(kb0 + 128) * ATT_DH, 0, 1);
    }

    // ---- stage K tile (128x64 halves): pure b128 copy ----
    for (int idx = tid; idx < 128 * 8; idx += 128) {
      const int row = idx >> 3;
      const int c   = (idx & 7) * 8;
      *(v8h*)&Klds[row * 72 + c] =
          *(const v8h*)(Kh + (size_t)(kb0 + row) * ATT_DH + c);
    }
    // ---- stage V tile transposed: Vtlds[d][key] ----
    for (int idx = tid; idx < 128 * 8; idx += 128) {
      const int key = idx >> 3;          // key
      const int c   = (idx & 7) * 8;     // d base
      const v8h v = *(const v8h*)(Vh + (size_t)(kb0 + key) * ATT_DH + c);
#pragma unroll
      for (int j = 0; j < 8; ++j) Vtlds[(c + j) * 136 + key] = v[j];
    }
    __syncthreads();

    // ---- scores: 16-row x 128-key tile = 8 frags, K-dim = 64 (2 steps) ----
    v8f sf[8];
#pragma unroll
    for (int ni = 0; ni < 8; ++ni) {
      v8f s = {};
#pragma unroll
      for (int ds = 0; ds < 2; ++ds) {
        const int krow = ni * 16 + l16;          // key (column of K^T)
        const int kb   = ds * 32 + hi * 8;       // d offset within K row
        const v16h bf = hcombine(*(const v8h*)&Klds[krow * 72 + kb],
                                 *(const v8h*)&Klds[krow * 72 + kb + 16]);
        s = wmma_f16(qf[ds], bf, s);
      }
      sf[ni] = s;
    }

    // ---- scale by 1/sqrt(64), mask (exactly like reference) ----
#pragma unroll
    for (int ni = 0; ni < 8; ++ni) {
      const int kcol = kb0 + ni * 16 + l16;
#pragma unroll
      for (int r = 0; r < 8; ++r) {
        const int qrow = q0 + wave * 16 + hi * 8 + r;
        const int mval = Mb[(size_t)qrow * ATT_L + kcol];
        const float sc = sf[ni][r] * 0.125f;
        sf[ni][r] = (mval == 0) ? -1.0e9f : sc;
      }
    }

    // ---- block row-max (reduce across the 16-lane group) ----
    float bmax[8];
#pragma unroll
    for (int r = 0; r < 8; ++r) {
      float m = sf[0][r];
#pragma unroll
      for (int ni = 1; ni < 8; ++ni) m = fmaxf(m, sf[ni][r]);
      bmax[r] = m;
    }
#pragma unroll
    for (int off = 1; off < 16; off <<= 1)
#pragma unroll
      for (int r = 0; r < 8; ++r)
        bmax[r] = fmaxf(bmax[r], __shfl_xor(bmax[r], off, 32));

    // ---- online softmax update ----
    float corr[8];
#pragma unroll
    for (int r = 0; r < 8; ++r) {
      const float mnew = fmaxf(rmax[r], bmax[r]);
      corr[r] = __expf(rmax[r] - mnew);
      rmax[r] = mnew;
    }

    float bsum[8] = {0.f, 0.f, 0.f, 0.f, 0.f, 0.f, 0.f, 0.f};
    _Float16* Pw = &Plds[wave * 16 * 136];
#pragma unroll
    for (int ni = 0; ni < 8; ++ni) {
#pragma unroll
      for (int r = 0; r < 8; ++r) {
        const float p = __expf(sf[ni][r] - rmax[r]);
        bsum[r] += p;
        Pw[(hi * 8 + r) * 136 + ni * 16 + l16] = (_Float16)p;
      }
    }
#pragma unroll
    for (int off = 1; off < 16; off <<= 1)
#pragma unroll
      for (int r = 0; r < 8; ++r)
        bsum[r] += __shfl_xor(bsum[r], off, 32);
#pragma unroll
    for (int r = 0; r < 8; ++r) rsum[r] = rsum[r] * corr[r] + bsum[r];

    // rescale running context
#pragma unroll
    for (int dt = 0; dt < 4; ++dt)
#pragma unroll
      for (int r = 0; r < 8; ++r) ctx[dt][r] *= corr[r];

    // ---- ctx += P @ V  (k-dim = 128 keys, 4 WMMA k-steps; d = 4 tiles) ----
#pragma unroll
    for (int ks = 0; ks < 4; ++ks) {
      const int kb = ks * 32 + hi * 8;
      const _Float16* pr = &Pw[l16 * 136];
      const v16h pfrag = hcombine(*(const v8h*)(pr + kb),
                                  *(const v8h*)(pr + kb + 16));
#pragma unroll
      for (int dt = 0; dt < 4; ++dt) {
        const int d   = dt * 16 + l16;
        const int vkb = ks * 32 + hi * 8;
        const v16h vfrag = hcombine(*(const v8h*)&Vtlds[d * 136 + vkb],
                                    *(const v8h*)&Vtlds[d * 136 + vkb + 16]);
        ctx[dt] = wmma_f16(pfrag, vfrag, ctx[dt]);
      }
    }
  }

  // ---- normalize and scatter (f16) to (b, l, h*64 + d) ----
  _Float16* outb = Ctx + (size_t)b * ATT_L * ATT_HID + (size_t)h * ATT_DH;
#pragma unroll
  for (int dt = 0; dt < 4; ++dt) {
#pragma unroll
    for (int r = 0; r < 8; ++r) {
      const int qrow = q0 + wave * 16 + hi * 8 + r;
      outb[(size_t)qrow * ATT_HID + dt * 16 + l16] =
          (_Float16)(ctx[dt][r] / rsum[r]);
    }
  }
}

// ---------------------------------------------------------------------------
// Host launcher.  d_in order: queries, keys, values, mask,
//                 Wq, bq, Wk, bk, Wv, bv, Wo, bo
// Workspace: Qp | Kp | Vp | Ctx  (each 4096*1024 f16 = 8 MiB; 32 MiB total)
// ---------------------------------------------------------------------------
extern "C" void kernel_launch(void* const* d_in, const int* in_sizes, int n_in,
                              void* d_out, int out_size, void* d_ws,
                              size_t ws_size, hipStream_t stream) {
  (void)in_sizes; (void)n_in; (void)out_size; (void)ws_size;

  const float* queries = (const float*)d_in[0];
  const float* keys    = (const float*)d_in[1];
  const float* values  = (const float*)d_in[2];
  const int*   mask    = (const int*)d_in[3];
  const float* Wq = (const float*)d_in[4];
  const float* bq = (const float*)d_in[5];
  const float* Wk = (const float*)d_in[6];
  const float* bk = (const float*)d_in[7];
  const float* Wv = (const float*)d_in[8];
  const float* bv = (const float*)d_in[9];
  const float* Wo = (const float*)d_in[10];
  const float* bo = (const float*)d_in[11];
  float* out = (float*)d_out;

  const size_t plane = (size_t)ATT_M * ATT_HID;  // 4096*1024 halves
  _Float16* Qp  = (_Float16*)d_ws;
  _Float16* Kp  = Qp + plane;
  _Float16* Vp  = Kp + plane;
  _Float16* Ctx = Vp + plane;

  const dim3 ggrid(GM / 128, GN / 128);          // (32, 8)
  gemm_bias_wmma<_Float16, false>
      <<<ggrid, 256, 0, stream>>>(queries, Wq, bq, Qp);
  gemm_bias_wmma<_Float16, false>
      <<<ggrid, 256, 0, stream>>>(keys, Wk, bk, Kp);
  gemm_bias_wmma<_Float16, false>
      <<<ggrid, 256, 0, stream>>>(values, Wv, bv, Vp);

  const dim3 agrid(ATT_L / 64, ATT_NH, ATT_B);   // (32, 16, 2)
  attn_flash_wmma<<<agrid, 128, 0, stream>>>(Qp, Kp, Vp, mask, Ctx);

  gemm_bias_wmma<float, true>
      <<<ggrid, 256, 0, stream>>>(Ctx, Wo, bo, out);
}
